// VQEmbedding_23038204575870
// MI455X (gfx1250) — matrix operand order repaired and checked by reading, not compiled, not run on hardware
//
#include <hip/hip_runtime.h>
#include <hip/hip_bf16.h>

typedef __attribute__((ext_vector_type(16))) __bf16 v16bf;
typedef __attribute__((ext_vector_type(8)))  __bf16 v8bf;
typedef __attribute__((ext_vector_type(8)))  float  v8f;
typedef __attribute__((ext_vector_type(4)))  unsigned int v4u;
typedef __attribute__((ext_vector_type(8)))  unsigned int v8u;

#define NROWS 65536      // 16 * 4096
#define DDIM  256
#define KCODES 1024
#define ROWS_PER_WAVE 16
#define WAVES_PER_BLOCK 8
#define ROWS_PER_BLOCK (ROWS_PER_WAVE * WAVES_PER_BLOCK)   // 128
#define NBLOCKS (NROWS / ROWS_PER_BLOCK)                   // 512
#define NPARTIALS (NBLOCKS * WAVES_PER_BLOCK)              // 4096
#define NTILES (KCODES / 16)                               // 64

// ---------------- workspace layout (bytes) ----------------
#define WS_PARTIALS_OFF 0          // float[4096]
#define WS_HNORM_OFF    16384      // float[1024]  0.5*||e_k||^2 (exact f32)
#define WS_CB_OFF       20480      // __bf16[1024*256]

// ============================================================
// TDM: issue a 16x256 bf16 tile load (8 KB) global -> LDS.
// D# per CDNA5 ISA. 2D tensor -> groups 2/3 NULL
// (2-operand asm form). Scalar (SGPR) operands; caller must be a
// uniform (single-wave) branch. Tracked by TENSORcnt.
// ============================================================
__device__ __forceinline__ void tdm_load_tile(const __bf16* gsrc, unsigned lds_off) {
  unsigned long long ga = (unsigned long long)(size_t)gsrc;
  v4u g0;
  g0[0] = 1u;                                    // count=1, is_restore=0, gather=0
  g0[1] = lds_off;                               // [63:32] lds_addr
  g0[2] = (unsigned)ga;                          // [95:64] global_addr lo
  g0[3] = (unsigned)((ga >> 32) & 0x01FFFFFFu)   // [120:96] global_addr hi
        | 0x80000000u;                           // [127:126] type=2 ("image")
  v8u g1;
  g1[0] = (1u << 16);            // workgroup_mask=0, data_size=1 (2 bytes), no flags
  g1[1] = ((unsigned)DDIM) << 16;        // [63:48] tensor_dim0 lo16 = 256
  g1[2] = ((unsigned)KCODES) << 16;      // [79:64] dim0 hi=0 | [95:80] tensor_dim1 lo16 = 1024
  g1[3] = ((unsigned)DDIM) << 16;        // [111:96] dim1 hi=0 | [127:112] tile_dim0 = 256
  g1[4] = 16u;                           // [143:128] tile_dim1 = 16, tile_dim2 = 0
  g1[5] = (unsigned)DDIM;                // [207:160] tensor_dim0_stride = 256 (lo32)
  g1[6] = 0u;                            // stride0 hi | tensor_dim1_stride lo (unused, 2D)
  g1[7] = 0u;
  asm volatile("tensor_load_to_lds %0, %1" :: "s"(g0), "s"(g1) : "memory");
}

// ============================================================
// Kernel 1: codebook f32 -> bf16 + half squared norms.
// One wave per codebook row. 128 blocks x 256 threads.
// ============================================================
__global__ __launch_bounds__(256) void vq_prep_kernel(
    const float* __restrict__ cb, __bf16* __restrict__ cb16,
    float* __restrict__ hnorm) {
  const int lane = threadIdx.x & 31;
  const int wave = threadIdx.x >> 5;
  const int k = blockIdx.x * WAVES_PER_BLOCK + wave;   // codebook row
  const float* row = cb + (size_t)k * DDIM + lane * 8;
  float4 x0 = *(const float4*)(row + 0);
  float4 x1 = *(const float4*)(row + 4);

  v8bf o;
  o[0] = (__bf16)x0.x; o[1] = (__bf16)x0.y; o[2] = (__bf16)x0.z; o[3] = (__bf16)x0.w;
  o[4] = (__bf16)x1.x; o[5] = (__bf16)x1.y; o[6] = (__bf16)x1.z; o[7] = (__bf16)x1.w;
  *(v8bf*)(cb16 + (size_t)k * DDIM + lane * 8) = o;

  float s = x0.x*x0.x + x0.y*x0.y + x0.z*x0.z + x0.w*x0.w
          + x1.x*x1.x + x1.y*x1.y + x1.z*x1.z + x1.w*x1.w;
  #pragma unroll
  for (int m = 16; m >= 1; m >>= 1) s += __shfl_xor(s, m, 32);
  if (lane == 0) hnorm[k] = 0.5f * s;
}

// ============================================================
// Kernel 2: distance GEMM via WMMA, TDM-staged + double-buffered
// codebook tiles, argmin, gather, loss partials.
// 512 blocks x 256 threads (8 waves). Wave owns 16 z-rows.
// ============================================================
__global__ __launch_bounds__(256) void vq_main_kernel(
    const float* __restrict__ z, const float* __restrict__ cb,
    const __bf16* __restrict__ cb16, const float* __restrict__ hnorm,
    float* __restrict__ partials,
    float* __restrict__ outQ, float* __restrict__ outIdxF) {
  __shared__ __bf16 s_tile[2][16 * DDIM];   // 2 x 8 KB double buffer
  __shared__ float  s_hnorm[KCODES];        // 4 KB

  const int tid  = threadIdx.x;
  const int lane = tid & 31;
  const int wave = tid >> 5;
  const int ncol = lane & 15;               // N column (codes) / M row (z) selector
  const int hsel = lane >> 4;               // lane half

  // stage all half-norms once per block
  #pragma unroll
  for (int i = 0; i < KCODES / 256; ++i) s_hnorm[tid + i * 256] = hnorm[tid + i * 256];

  const size_t rowBase = (size_t)blockIdx.x * ROWS_PER_BLOCK + (size_t)wave * ROWS_PER_WAVE;
  const size_t myRow   = rowBase + ncol;    // this lane's z row for the A matrix

  // prologue: TDM tile 0 -> buffer 0 (wave 0 only; uniform scalar branch)
  if (wave == 0)
    tdm_load_tile(cb16, (unsigned)(size_t)(void*)&s_tile[0][0]);

  // ---- load A fragments (16x32 bf16 each), ISA §7.12.2 layout:
  // lanes 0-15 (hsel=0): halves 0-7 -> K d0+0..7,  halves 8-15 -> K d0+16..23
  // lanes 16-31(hsel=1): halves 0-7 -> K d0+8..15, halves 8-15 -> K d0+24..31
  v16bf a[8];
  {
    const float* zr = z + myRow * DDIM;
    #pragma unroll
    for (int f = 0; f < 8; ++f) {
      const int d0 = f * 32 + hsel * 8;
      float4 x0 = *(const float4*)(zr + d0);
      float4 x1 = *(const float4*)(zr + d0 + 4);
      float4 y0 = *(const float4*)(zr + d0 + 16);
      float4 y1 = *(const float4*)(zr + d0 + 20);
      a[f][0]  = (__bf16)x0.x; a[f][1]  = (__bf16)x0.y; a[f][2]  = (__bf16)x0.z; a[f][3]  = (__bf16)x0.w;
      a[f][4]  = (__bf16)x1.x; a[f][5]  = (__bf16)x1.y; a[f][6]  = (__bf16)x1.z; a[f][7]  = (__bf16)x1.w;
      a[f][8]  = (__bf16)y0.x; a[f][9]  = (__bf16)y0.y; a[f][10] = (__bf16)y0.z; a[f][11] = (__bf16)y0.w;
      a[f][12] = (__bf16)y1.x; a[f][13] = (__bf16)y1.y; a[f][14] = (__bf16)y1.z; a[f][15] = (__bf16)y1.w;
    }
  }

  float minv[8];
  int   mini[8];
  #pragma unroll
  for (int r = 0; r < 8; ++r) { minv[r] = 3.4e38f; mini[r] = 0; }

  for (int t = 0; t < NTILES; ++t) {
    // tile t is the only outstanding TDM op of wave 0 -> wait to 0.
    if (wave == 0) __builtin_amdgcn_s_wait_tensorcnt(0);
    // one barrier/iter: publishes tile t AND retires all reads of the
    // buffer that tile t+1 (issued below) will overwrite.
    __syncthreads();
    if (wave == 0 && t + 1 < NTILES)
      tdm_load_tile(cb16 + (size_t)(t + 1) * 16 * DDIM,
                    (unsigned)(size_t)(void*)&s_tile[(t + 1) & 1][0]);

    const __bf16* tile = s_tile[t & 1];
    v8f c = {0.f, 0.f, 0.f, 0.f, 0.f, 0.f, 0.f, 0.f};
    #pragma unroll
    for (int f = 0; f < 8; ++f) {
      // B fragment 32x16 bf16: lane ncol = code, same K order as A:
      // halves 0-7 -> K hsel*8+0..7, halves 8-15 -> K 16+hsel*8+0..7.
      v8bf b0 = *(const v8bf*)&tile[ncol * DDIM + f * 32 + hsel * 8];
      v8bf b1 = *(const v8bf*)&tile[ncol * DDIM + f * 32 + 16 + hsel * 8];
      v16bf b = __builtin_shufflevector(b0, b1, 0,1,2,3,4,5,6,7,8,9,10,11,12,13,14,15);
      c = __builtin_amdgcn_wmma_f32_16x16x32_bf16(false, a[f], false, b,
                                                  (short)0, c, false, false);
    }

    const int n = t * 16 + ncol;          // global code index of this lane's column
    const float hn = s_hnorm[n];
    // C/D layout: lanes 0-15 -> (M=r, N=lane); lanes 16-31 -> (M=r+8, N=lane-16)
    #pragma unroll
    for (int r = 0; r < 8; ++r) {
      float val = hn - c[r];              // 0.5||e||^2 - z.e  (monotone in distance)
      if (val < minv[r]) { minv[r] = val; mini[r] = n; }
    }
  }

  // reduce (min, argmin) across the 16 lanes sharing each M row (xor 1..8 stays in half)
  #pragma unroll
  for (int r = 0; r < 8; ++r) {
    float v = minv[r]; int i = mini[r];
    #pragma unroll
    for (int m = 8; m >= 1; m >>= 1) {
      float ov = __shfl_xor(v, m, 32);
      int   oi = __shfl_xor(i, m, 32);
      if (ov < v || (ov == v && oi < i)) { v = ov; i = oi; }
    }
    minv[r] = v; mini[r] = i;
  }

  // ---- output phase: gather f32 codebook rows, write quantized + index, loss partial
  float lossAcc = 0.f;
  #pragma unroll
  for (int m = 0; m < 16; ++m) {
    const int idx = (m < 8) ? __shfl(mini[m], 0, 32) : __shfl(mini[m - 8], 16, 32);
    const size_t row = rowBase + m;
    const float* qp = cb + (size_t)idx * DDIM + lane * 8;
    const float* zp = z + row * DDIM + lane * 8;
    float4 q0 = *(const float4*)(qp + 0);
    float4 q1 = *(const float4*)(qp + 4);
    float4 z0 = *(const float4*)(zp + 0);
    float4 z1 = *(const float4*)(zp + 4);
    float* op = outQ + row * DDIM + lane * 8;
    *(float4*)(op + 0) = q0;
    *(float4*)(op + 4) = q1;
    float d;
    d = q0.x - z0.x; lossAcc += d * d;  d = q0.y - z0.y; lossAcc += d * d;
    d = q0.z - z0.z; lossAcc += d * d;  d = q0.w - z0.w; lossAcc += d * d;
    d = q1.x - z1.x; lossAcc += d * d;  d = q1.y - z1.y; lossAcc += d * d;
    d = q1.z - z1.z; lossAcc += d * d;  d = q1.w - z1.w; lossAcc += d * d;
    if (lane == 0) outIdxF[row] = (float)idx;
  }
  #pragma unroll
  for (int m = 16; m >= 1; m >>= 1) lossAcc += __shfl_xor(lossAcc, m, 32);
  if (lane == 0) partials[blockIdx.x * WAVES_PER_BLOCK + wave] = lossAcc;
}

// ============================================================
// Kernel 3: deterministic fixed-order reduction of loss partials.
// loss = (1 + 0.25) * sum((q - z)^2) / (N * D)
// ============================================================
__global__ __launch_bounds__(256) void vq_finalize_kernel(
    const float* __restrict__ partials, float* __restrict__ outLoss) {
  __shared__ float s[256];
  float acc = 0.f;
  for (int i = threadIdx.x; i < NPARTIALS; i += 256) acc += partials[i];
  s[threadIdx.x] = acc;
  __syncthreads();
  for (int st = 128; st > 0; st >>= 1) {
    if (threadIdx.x < st) s[threadIdx.x] += s[threadIdx.x + st];
    __syncthreads();
  }
  if (threadIdx.x == 0)
    *outLoss = 1.25f * s[0] / (float)((size_t)NROWS * DDIM);
}

extern "C" void kernel_launch(void* const* d_in, const int* in_sizes, int n_in,
                              void* d_out, int out_size, void* d_ws, size_t ws_size,
                              hipStream_t stream) {
  (void)in_sizes; (void)n_in; (void)out_size; (void)ws_size;
  const float* z  = (const float*)d_in[0];
  const float* cb = (const float*)d_in[1];

  char* ws = (char*)d_ws;
  float*  ws_partials = (float*)(ws + WS_PARTIALS_OFF);
  float*  ws_hnorm    = (float*)(ws + WS_HNORM_OFF);
  __bf16* ws_cb16     = (__bf16*)(ws + WS_CB_OFF);

  float* outAll  = (float*)d_out;
  float* outQ    = outAll;                                    // [N, D] quantized_st
  float* outLoss = outAll + (size_t)NROWS * DDIM;             // scalar loss
  float* outIdxF = outAll + (size_t)NROWS * DDIM + 1;         // [N] indices (as float)

  vq_prep_kernel<<<KCODES / WAVES_PER_BLOCK, 256, 0, stream>>>(cb, ws_cb16, ws_hnorm);
  vq_main_kernel<<<NBLOCKS, 256, 0, stream>>>(z, cb, ws_cb16, ws_hnorm,
                                              ws_partials, outQ, outIdxF);
  vq_finalize_kernel<<<1, 256, 0, stream>>>(ws_partials, outLoss);
}
